// TFNInteractionBlock_51058571215444
// MI455X (gfx1250) — matrix operand order repaired and checked by reading, not compiled, not run on hardware
//
#include <hip/hip_runtime.h>

#define N_NODES 10000
#define E_EDGES 160000
#define NB 16
#define H_DIM 64

typedef __attribute__((ext_vector_type(16))) _Float16 v16h;
typedef __attribute__((ext_vector_type(8)))  _Float16 v8h;
typedef __attribute__((ext_vector_type(8)))  float    v8f;

#define ALPHA   0.17677669529663687f   // 1/sqrt(2*MUL) = 1/sqrt(32)
#define INVSQ3  0.57735026918962576f   // 1/sqrt(3)
#define SCALE_S 0.25f                  // 1/sqrt(MUL)
#define EPS     1e-8f

__device__ __forceinline__ float sigmoidf_(float x) { return 1.0f / (1.0f + __expf(-x)); }

// ---------------------------------------------------------------------------
// Workspace layout:
//   [0, N*16) floats          agg_s
//   [N*16, N*64) floats       agg_v  ([n][u][i], i fastest)
//   byte off N*64*4           w2 f16 B-fragments: [kt(2)][tile_col(64)][lane(32)][16 f16]
// ---------------------------------------------------------------------------

__global__ void zero_agg_kernel(float* __restrict__ agg) {
    int i = blockIdx.x * blockDim.x + threadIdx.x;
    if (i < N_NODES * 64) agg[i] = 0.0f;
}

// Pre-swizzle w2 (64x1024 f32, row-major) into WMMA B-fragment layout, f16.
// B (32x16 f16): lanes 0-15 hold K=kt*32+{0..7,16..23} of column n=lane;
//                lanes 16-31 hold K=kt*32+{8..15,24..31} of column n=lane-16.
__global__ void prep_w2_kernel(const float* __restrict__ w2, _Float16* __restrict__ w2f) {
    int t = blockIdx.x * blockDim.x + threadIdx.x;   // 4096 threads
    int kt       = t >> 11;
    int tile_col = (t >> 5) & 63;
    int lane     = t & 31;
    int n     = tile_col * 16 + (lane & 15);
    int kbase = kt * 32 + ((lane < 16) ? 0 : 8);
    _Float16* dst = w2f + (long)t * 16;
#pragma unroll
    for (int j = 0; j < 16; ++j)
        dst[j] = (_Float16)w2[(long)(kbase + (j < 8 ? j : j + 8)) * 1024 + n];
}

// ---------------------------------------------------------------------------
// Edge kernel: one block = 32 edges (two 16-edge M-subtiles), 256 thr = 8 waves.
// Wave w owns output columns [w*128, w*128+128): idx = w/2, u = (w%2)*8 + t.
// ---------------------------------------------------------------------------
__global__ __launch_bounds__(256)
__attribute__((amdgpu_waves_per_eu(4)))
void edge_kernel(const float* __restrict__ node_features,
                 const float* __restrict__ edge_sh,
                 const float* __restrict__ edge_radial,
                 const float* __restrict__ w1,
                 const float* __restrict__ b1,
                 const float* __restrict__ b2,
                 const int*   __restrict__ edge_index,
                 const _Float16* __restrict__ w2f,
                 float* __restrict__ agg_s,
                 float* __restrict__ agg_v)
{
    __shared__ __align__(16) _Float16 sh_h[32][64];     // silu(r@w1+b1), f16
    // coefficient tables transposed to [u][e] for vectorized v8f LDS loads
    __shared__ __align__(32) float sc_xssh[16][32];     // xs*shs
    __shared__ __align__(32) float sc_xs  [16][32];     // xs
    __shared__ __align__(32) float sc_xvd [16][32];     // (1/sqrt3)*(xv . shv)
    __shared__ __align__(32) float sc_xvsh[3][16][32];  // xv_i*shs (channel stride 512)
    __shared__ float sh_shs[32];
    __shared__ float sh_shv[32][3];
    __shared__ int   sh_src[32];
    __shared__ int   sh_dst[32];
    __shared__ float sh_msgs[32][16];
    __shared__ float sh_t01[32][16];
    __shared__ float sh_t10[32][16][3];

    const int tid = threadIdx.x;
    const int e0  = blockIdx.x * 32;

    if (tid < 32) {
        int e = e0 + tid;
        sh_src[tid] = edge_index[e];
        sh_dst[tid] = edge_index[E_EDGES + e];
        sh_shs[tid]    = edge_sh[(long)e * 4 + 0];
        sh_shv[tid][0] = edge_sh[(long)e * 4 + 1];
        sh_shv[tid][1] = edge_sh[(long)e * 4 + 2];
        sh_shv[tid][2] = edge_sh[(long)e * 4 + 3];
    }
    {   // zero the message accumulators (512 + 512 + 1536 floats)
        ((float*)sh_msgs)[tid]       = 0.0f;
        ((float*)sh_msgs)[tid + 256] = 0.0f;
        ((float*)sh_t01)[tid]        = 0.0f;
        ((float*)sh_t01)[tid + 256]  = 0.0f;
#pragma unroll
        for (int k = 0; k < 6; ++k) ((float*)sh_t10)[tid + k * 256] = 0.0f;
    }
    __syncthreads();

    // Per-(edge,u) coefficient tables from gathered source features
#pragma unroll
    for (int p = tid; p < 512; p += 256) {
        int e = p >> 4, u = p & 15;
        const float* nf = node_features + (long)sh_src[e] * 64;
        float xs = nf[u];
        float x0 = nf[16 + u * 3 + 0];
        float x1 = nf[16 + u * 3 + 1];
        float x2 = nf[16 + u * 3 + 2];
        float ss = sh_shs[e];
        sc_xs[u][e]   = xs;
        sc_xssh[u][e] = xs * ss;
        sc_xvd[u][e]  = INVSQ3 * (x0 * sh_shv[e][0] + x1 * sh_shv[e][1] + x2 * sh_shv[e][2]);
        sc_xvsh[0][u][e] = x0 * ss;
        sc_xvsh[1][u][e] = x1 * ss;
        sc_xvsh[2][u][e] = x2 * ss;
    }
    // h = silu(r @ w1 + b1): thread (e = tid&31, wave-slot) computes 8 entries
    {
        int e  = tid & 31;
        int kb = (tid >> 5) * 8;
        const float* r = edge_radial + (long)(e0 + e) * NB;
        float rv[NB];
#pragma unroll
        for (int j = 0; j < NB; ++j) rv[j] = r[j];
#pragma unroll
        for (int kk = 0; kk < 8; ++kk) {
            int k = kb + kk;
            float acc = b1[k];
#pragma unroll
            for (int j = 0; j < NB; ++j) acc += rv[j] * w1[j * H_DIM + k];
            sh_h[e][k] = (_Float16)(acc * sigmoidf_(acc));
        }
    }
    __syncthreads();

    const int lane = tid & 31;
    const int wave = __builtin_amdgcn_readfirstlane(tid >> 5);  // scalar -> s_cbranch
    const int idx  = wave >> 1;                                  // W block (0..3)
    const int hl   = lane >> 4;
    const int eb   = hl * 8;           // C-fragment row base within subtile
    const int vcol = lane & 15;
    const int arow = lane & 15;
    const int koff = hl ? 8 : 0;

    // A-fragments (16x32 f16) for both subtiles, both k-tiles
    v16h af[2][2];
#pragma unroll
    for (int s = 0; s < 2; ++s) {
        const int er = s * 16 + arow;
        v8h lo0 = *(const v8h*)&sh_h[er][ 0 + koff];
        v8h hi0 = *(const v8h*)&sh_h[er][16 + koff];
        v8h lo1 = *(const v8h*)&sh_h[er][32 + koff];
        v8h hi1 = *(const v8h*)&sh_h[er][48 + koff];
        af[s][0] = __builtin_shufflevector(lo0, hi0, 0,1,2,3,4,5,6,7,8,9,10,11,12,13,14,15);
        af[s][1] = __builtin_shufflevector(lo1, hi1, 0,1,2,3,4,5,6,7,8,9,10,11,12,13,14,15);
    }

    // Channel-0 table selected scalar-ly (every wave runs the 1-channel path;
    // only idx==2 waves additionally accumulate channels 1 and 2).
    const float* tbl = (idx == 0) ? &sc_xssh[0][0]
                     : (idx == 1) ? &sc_xs[0][0]
                     : (idx == 3) ? &sc_xvd[0][0]
                                  : &sc_xvsh[0][0][0];

    float pa[2][8], pb[2][8], pc[2][8];
#pragma unroll
    for (int s = 0; s < 2; ++s)
#pragma unroll
        for (int r = 0; r < 8; ++r) { pa[s][r] = 0.f; pb[s][r] = 0.f; pc[s][r] = 0.f; }

#pragma unroll 2
    for (int t = 0; t < 8; ++t) {
        const int tile_col = wave * 8 + t;
        const int u = tile_col & 15;
        v16h bf0 = *(const v16h*)(w2f + ((long)(     tile_col) * 32 + lane) * 16);
        v16h bf1 = *(const v16h*)(w2f + ((long)(64 + tile_col) * 32 + lane) * 16);
        v8f acc0 = {}, acc1 = {};
        acc0 = __builtin_amdgcn_wmma_f32_16x16x32_f16(false, af[0][0], false, bf0, (short)0, acc0, false, false);
        acc0 = __builtin_amdgcn_wmma_f32_16x16x32_f16(false, af[0][1], false, bf1, (short)0, acc0, false, false);
        acc1 = __builtin_amdgcn_wmma_f32_16x16x32_f16(false, af[1][0], false, bf0, (short)0, acc1, false, false);
        acc1 = __builtin_amdgcn_wmma_f32_16x16x32_f16(false, af[1][1], false, bf1, (short)0, acc1, false, false);
        const float b2v = b2[tile_col * 16 + vcol];
        const float* trow = tbl + u * 32 + eb;
#pragma unroll
        for (int s = 0; s < 2; ++s) {
            const v8f c0 = *(const v8f*)(trow + s * 16);
            const v8f* ac = s ? &acc1 : &acc0;
#pragma unroll
            for (int r = 0; r < 8; ++r)
                pa[s][r] += c0[r] * ((*ac)[r] + b2v);
        }
        if (idx == 2) {
#pragma unroll
            for (int s = 0; s < 2; ++s) {
                const v8f c1 = *(const v8f*)(trow +  512 + s * 16);
                const v8f c2 = *(const v8f*)(trow + 1024 + s * 16);
                const v8f* ac = s ? &acc1 : &acc0;
#pragma unroll
                for (int r = 0; r < 8; ++r) {
                    float Wv = (*ac)[r] + b2v;
                    pb[s][r] += c1[r] * Wv;
                    pc[s][r] += c2[r] * Wv;
                }
            }
        }
    }

    if (idx == 2) {
#pragma unroll
        for (int s = 0; s < 2; ++s)
#pragma unroll
            for (int r = 0; r < 8; ++r) {
                const int m = s * 16 + eb + r;
                atomicAdd(&sh_t10[m][vcol][0], pa[s][r]);
                atomicAdd(&sh_t10[m][vcol][1], pb[s][r]);
                atomicAdd(&sh_t10[m][vcol][2], pc[s][r]);
            }
    } else {
        float* dst = (idx == 1) ? &sh_t01[0][0] : &sh_msgs[0][0];
#pragma unroll
        for (int s = 0; s < 2; ++s)
#pragma unroll
            for (int r = 0; r < 8; ++r) {
                const int m = s * 16 + eb + r;
                atomicAdd(&dst[m * 16 + vcol], pa[s][r]);
            }
    }
    __syncthreads();

    // Scatter: global f32 atomics into agg buffers
    for (int p = tid; p < 512; p += 256) {
        int e = p >> 4, vv = p & 15;
        int d = sh_dst[e];
        unsafeAtomicAdd(&agg_s[(long)d * 16 + vv], ALPHA * sh_msgs[e][vv]);
        float tb = sh_t01[e][vv];
#pragma unroll
        for (int i = 0; i < 3; ++i) {
            float mv = ALPHA * (tb * sh_shv[e][i] + sh_t10[e][vv][i]);
            unsafeAtomicAdd(&agg_v[((long)d * 16 + vv) * 3 + i], mv);
        }
    }
}

// ---------------------------------------------------------------------------
// Node kernel: per-(node, v) thread — 16x16 linears, norm gates, residual.
// ---------------------------------------------------------------------------
__global__ __launch_bounds__(256)
void node_kernel(const float* __restrict__ nf,
                 const float* __restrict__ agg_s,
                 const float* __restrict__ agg_v,
                 const float* __restrict__ lw0,
                 const float* __restrict__ lw1,
                 float* __restrict__ out)
{
    int t = blockIdx.x * 256 + threadIdx.x;
    if (t >= N_NODES * 16) return;
    int n = t >> 4, v = t & 15;
    const float* as = agg_s + (long)n * 16;
    const float* av = agg_v + (long)n * 48;
    float ts = 0.f, t0 = 0.f, t1 = 0.f, t2 = 0.f;
#pragma unroll
    for (int u = 0; u < 16; ++u) {
        ts += as[u] * lw0[u * 16 + v];
        float w = lw1[u * 16 + v];
        t0 += av[u * 3 + 0] * w;
        t1 += av[u * 3 + 1] * w;
        t2 += av[u * 3 + 2] * w;
    }
    ts *= SCALE_S; t0 *= SCALE_S; t1 *= SCALE_S; t2 *= SCALE_S;
    float ns = fabsf(ts);
    float gs = (ns * sigmoidf_(ns)) / (ns + EPS);
    float nv = sqrtf(t0 * t0 + t1 * t1 + t2 * t2);
    float gv = (nv * sigmoidf_(nv)) / (nv + EPS);
    const float* nfr = nf + (long)n * 64;
    float* o = out + (long)n * 64;
    o[v] = nfr[v] + ts * gs;
    o[16 + v * 3 + 0] = nfr[16 + v * 3 + 0] + t0 * gv;
    o[16 + v * 3 + 1] = nfr[16 + v * 3 + 1] + t1 * gv;
    o[16 + v * 3 + 2] = nfr[16 + v * 3 + 2] + t2 * gv;
}

extern "C" void kernel_launch(void* const* d_in, const int* in_sizes, int n_in,
                              void* d_out, int out_size, void* d_ws, size_t ws_size,
                              hipStream_t stream)
{
    const float* node_features = (const float*)d_in[0];
    const float* edge_sh       = (const float*)d_in[1];
    const float* edge_radial   = (const float*)d_in[2];
    const float* w1  = (const float*)d_in[3];
    const float* b1  = (const float*)d_in[4];
    const float* w2  = (const float*)d_in[5];
    const float* b2  = (const float*)d_in[6];
    const float* lw0 = (const float*)d_in[7];
    const float* lw1 = (const float*)d_in[8];
    const int* edge_index = (const int*)d_in[9];

    float* agg   = (float*)d_ws;                    // N*64 floats
    float* agg_s = agg;
    float* agg_v = agg + (long)N_NODES * 16;
    _Float16* w2f = (_Float16*)((char*)d_ws + (size_t)N_NODES * 64 * 4);  // 128 KB

    zero_agg_kernel<<<(N_NODES * 64 + 255) / 256, 256, 0, stream>>>(agg);
    prep_w2_kernel<<<16, 256, 0, stream>>>(w2, w2f);
    edge_kernel<<<E_EDGES / 32, 256, 0, stream>>>(node_features, edge_sh, edge_radial,
                                                  w1, b1, b2, edge_index, w2f,
                                                  agg_s, agg_v);
    node_kernel<<<(N_NODES * 16 + 255) / 256, 256, 0, stream>>>(node_features, agg_s, agg_v,
                                                                lw0, lw1, (float*)d_out);
}
